// WaveletDenoising_67860483277348
// MI455X (gfx1250) — matrix operand order repaired and checked by reading, not compiled, not run on hardware
//
#include <hip/hip_runtime.h>

typedef __attribute__((ext_vector_type(16))) _Float16 v16h;
typedef __attribute__((ext_vector_type(8)))  float    v8f;

#define BATCH 16
#define NPIX  512
#define NHALF 256
#define NQUAR 128
#define PITCH 17
#define S1 (BATCH*NHALF*NHALF)                 /* 1,048,576 floats per band per channel (L1) */
#define S2 (BATCH*NQUAR*NQUAR)                 /* 262,144 floats per band per channel (L2)  */
#define OFF_L2  ((size_t)12*S1)
#define OFF_SIG ((size_t)12*S1 + (size_t)12*S2)
#define OFF_THR (OFF_SIG + 48)

// ---------------- Haar matrix (forward A, inverse A^T) ----------------
static __device__ __forceinline__ float haarM(int m, int k, int inv) {
  if (inv) { int t = m; m = k; k = t; }
  const float h = 0.70710678118654752440f;
  if (m < 8) return (k == 2*m || k == 2*m + 1) ? h : 0.0f;
  int mm = m - 8;
  if (k == 2*mm)     return  h;
  if (k == 2*mm + 1) return -h;
  return 0.0f;
}

// A-operand (16x32 f16) lane layout: lane L holds M=L%16;
// lanes<16: halves 0-15 => K=h; lanes>=16: K=16+h (contiguous, matches B).
// a1 = [M_hi | M_hi], a2 = [M_lo | 0]  (split-precision f16 pair)
static __device__ __forceinline__ void build_a(int lane, int inv, v16h& a1, v16h& a2) {
  int m  = lane & 15;
  int kb = (lane < 16) ? 0 : 16;
  #pragma unroll
  for (int h = 0; h < 16; ++h) {
    int k32 = kb + h;
    float v = haarM(m, k32 & 15, inv);
    _Float16 vh = (_Float16)v;
    a1[h] = vh;
    a2[h] = (k32 < 16) ? (_Float16)(v - (float)vh) : (_Float16)0.0f;
  }
}

// B-operand (32x16 f16): lanes<16 hold K=0..15 (N=lane), lanes>=16 hold K=16..31 (N=lane-16).
// b1 = [X_hi ; X_lo], b2 = [X_hi ; 0]
static __device__ __forceinline__ void build_b(const float* p, int sh, int si, int lane,
                                               v16h& b1, v16h& b2) {
  int idx = lane & 15;
  int hiHalf = (lane < 16);
  #pragma unroll
  for (int h = 0; h < 16; ++h) {
    float v = p[h*sh + idx*si];
    _Float16 vh = (_Float16)v;
    _Float16 vl = (_Float16)(v - (float)vh);
    b1[h] = hiHalf ? vh : vl;
    b2[h] = hiHalf ? vh : (_Float16)0.0f;
  }
}

// R = M * X * M^T on a 16x16 LDS tile. Returns lane layout:
// lane L holds R[L%16][8*(L/16) + r], r=0..7  (row L%16, 8 contiguous columns).
static __device__ __forceinline__ v8f tile_mxmt(const float* xs, float* ts, int lane,
                                                v16h a1, v16h a2) {
  v16h b1, b2;
  build_b(xs, PITCH, 1, lane, b1, b2);  // columns of X -> B = [X_hi; X_lo]
  v8f t = {};
  t = __builtin_amdgcn_wmma_f32_16x16x32_f16(false, a1, false, b1, (short)0, t, false, false);
  t = __builtin_amdgcn_wmma_f32_16x16x32_f16(false, a2, false, b2, (short)0, t, false, false);
  // T is in D-layout: (vgpr r, lane) = T[r + 8*(lane/16)][lane%16]; stash row-major in LDS.
  int nn = lane & 15, mb = (lane >> 4) << 3;
  #pragma unroll
  for (int r = 0; r < 8; ++r) ts[(mb + r)*PITCH + nn] = t[r];
  asm volatile("s_wait_dscnt 0x0" ::: "memory");
  build_b(ts, 1, PITCH, lane, b1, b2);  // rows of T -> B = T^T split; compute R^T = M * T^T
  v8f res = {};
  res = __builtin_amdgcn_wmma_f32_16x16x32_f16(false, a1, false, b1, (short)0, res, false, false);
  res = __builtin_amdgcn_wmma_f32_16x16x32_f16(false, a2, false, b2, (short)0, res, false, false);
  return res;
}

static __device__ __forceinline__ float soft_thr(float v, float t) {
  float a = fabsf(v) - t;
  a = (a > 0.0f) ? a : 0.0f;
  return copysignf(a, v);
}

// ---------------- Pass 1: RGB->YCbCr + level-1 DWT ----------------
__global__ __launch_bounds__(128) void k_fwd1(const float* __restrict__ img, float* __restrict__ ws) {
  __shared__ float sh[4][4*16*PITCH];
  int wave = threadIdx.x >> 5, lane = threadIdx.x & 31;
  int tile = blockIdx.x*4 + wave;                 // 16384 tiles
  int b = tile >> 10, tI = (tile >> 5) & 31, tJ = tile & 31;
  float* xs = &sh[wave][0];
  float* ts = &sh[wave][3*16*PITCH];
  const float* src = img + (size_t)b*NPIX*NPIX*3;
  #pragma unroll
  for (int i2 = 0; i2 < 8; ++i2) {
    int q = lane + 32*i2, r = q >> 4, c = q & 15;
    size_t p = ((size_t)(tI*16 + r)*NPIX + (size_t)(tJ*16 + c))*3;
    float R = src[p], G = src[p+1], Bl = src[p+2];
    float y  = 0.299f*R + 0.587f*G + 0.114f*Bl;
    float cb = (Bl - y)*0.564f + 0.5f;
    float cr = (R  - y)*0.713f + 0.5f;
    xs[0*16*PITCH + r*PITCH + c] = y;
    xs[1*16*PITCH + r*PITCH + c] = cb;
    xs[2*16*PITCH + r*PITCH + c] = cr;
  }
  asm volatile("s_wait_dscnt 0x0" ::: "memory");
  v16h a1, a2; build_a(lane, 0, a1, a2);
  int i = lane & 15, mb = (lane >> 4) << 3;
  int region = (i >= 8) ? ((mb >= 8) ? 3 : 1) : ((mb >= 8) ? 2 : 0);  // ll,lh,hl,hh
  #pragma unroll
  for (int ch = 0; ch < 3; ++ch) {
    v8f rr = tile_mxmt(xs + ch*16*PITCH, ts, lane, a1, a2);
    float* dp = ws + (size_t)region*3*S1 + (size_t)ch*S1 + (size_t)b*NHALF*NHALF
                   + (size_t)(tI*8 + (i & 7))*NHALF + (size_t)tJ*8;
    float4* d4 = (float4*)dp;
    d4[0] = make_float4(rr[0], rr[1], rr[2], rr[3]);
    d4[1] = make_float4(rr[4], rr[5], rr[6], rr[7]);
  }
}

// ---------------- Pass 2: level-2 DWT on ll1 ----------------
__global__ __launch_bounds__(128) void k_fwd2(float* __restrict__ ws) {
  __shared__ float sh[4][2*16*PITCH];
  int wave = threadIdx.x >> 5, lane = threadIdx.x & 31;
  int tile = blockIdx.x*4 + wave;                 // 12288 tiles
  int ch = tile >> 12, rem = tile & 4095;
  int b = rem >> 8, tI = (rem >> 4) & 15, tJ = rem & 15;
  float* xs = &sh[wave][0];
  float* ts = &sh[wave][16*PITCH];
  const float* src = ws + (size_t)ch*S1 + (size_t)b*NHALF*NHALF;
  #pragma unroll
  for (int i2 = 0; i2 < 8; ++i2) {
    int q = lane + 32*i2, r = q >> 4, c = q & 15;
    xs[r*PITCH + c] = src[(size_t)(tI*16 + r)*NHALF + (size_t)(tJ*16 + c)];
  }
  asm volatile("s_wait_dscnt 0x0" ::: "memory");
  v16h a1, a2; build_a(lane, 0, a1, a2);
  v8f rr = tile_mxmt(xs, ts, lane, a1, a2);
  int i = lane & 15, mb = (lane >> 4) << 3;
  int region = (i >= 8) ? ((mb >= 8) ? 3 : 1) : ((mb >= 8) ? 2 : 0);
  float* dp = ws + OFF_L2 + (size_t)region*3*S2 + (size_t)ch*S2 + (size_t)b*NQUAR*NQUAR
                 + (size_t)(tI*8 + (i & 7))*NQUAR + (size_t)tJ*8;
  float4* d4 = (float4*)dp;
  d4[0] = make_float4(rr[0], rr[1], rr[2], rr[3]);
  d4[1] = make_float4(rr[4], rr[5], rr[6], rr[7]);
}

// ---------------- Median / MAD via exact radix select ----------------
static __device__ __forceinline__ unsigned fkey(float f) {
  unsigned u = __float_as_uint(f);
  return (u & 0x80000000u) ? ~u : (u | 0x80000000u);
}
static __device__ __forceinline__ float keyf(unsigned k) {
  unsigned u = (k & 0x80000000u) ? (k & 0x7FFFFFFFu) : ~k;
  return __uint_as_float(u);
}

static __device__ float radix_select(const float* d, int n, unsigned k, float center, int absMode) {
  __shared__ unsigned hist[256];
  __shared__ unsigned sP, sK;
  unsigned pfx = 0, kk = k;
  for (int pass = 0; pass < 4; ++pass) {
    int shift = 24 - 8*pass;
    hist[threadIdx.x] = 0u;
    __syncthreads();
    unsigned maskHigh = (pass == 0) ? 0u : (0xFFFFFFFFu << (shift + 8));
    for (int i = threadIdx.x; i < n; i += 256) {
      float v = d[i];
      if (absMode) v = fabsf(v - center);
      unsigned u = fkey(v);
      if ((u & maskHigh) == pfx) atomicAdd(&hist[(u >> shift) & 255u], 1u);
    }
    __syncthreads();
    if (threadIdx.x == 0) {
      unsigned cum = 0, bb = 0;
      for (bb = 0; bb < 256; ++bb) { unsigned c = hist[bb]; if (cum + c > kk) break; cum += c; }
      if (bb > 255u) bb = 255u;
      sP = pfx | (bb << shift);
      sK = kk - cum;
    }
    __syncthreads();
    pfx = sP; kk = sK;
    __syncthreads();
  }
  return keyf(pfx);
}

__global__ __launch_bounds__(256) void k_median(float* __restrict__ ws) {
  int g = blockIdx.x;                       // 48 groups: ch*16 + image
  int ch = g >> 4, b = g & 15;
  const float* d = ws + (size_t)9*S1 + (size_t)ch*S1 + (size_t)b*(NHALF*NHALF);  // hh1
  const int n = NHALF*NHALF;
  float m1 = radix_select(d, n, (unsigned)(n/2 - 1), 0.0f, 0);
  float m2 = radix_select(d, n, (unsigned)(n/2),     0.0f, 0);
  float med = 0.5f*(m1 + m2);
  float q1 = radix_select(d, n, (unsigned)(n/2 - 1), med, 1);
  float q2 = radix_select(d, n, (unsigned)(n/2),     med, 1);
  if (threadIdx.x == 0) ws[OFF_SIG + g] = (0.5f*(q1 + q2)) / 0.6745f;
}

// ---------------- Variance + BayesShrink thresholds ----------------
__global__ __launch_bounds__(256) void k_var(float* __restrict__ ws) {
  int id = blockIdx.x;                      // 288 = 48 groups * 6 bands
  int band = id % 6, g = id / 6, ch = g >> 4, b = g & 15;
  const float* d; int n;
  if (band < 3) { n = NHALF*NHALF; d = ws + (size_t)(band+1)*3*S1 + (size_t)ch*S1 + (size_t)b*n; }
  else          { n = NQUAR*NQUAR; d = ws + OFF_L2 + (size_t)(band-2)*3*S2 + (size_t)ch*S2 + (size_t)b*n; }
  float s = 0.0f, s2 = 0.0f;
  for (int i = threadIdx.x; i < n; i += 256) { float v = d[i]; s += v; s2 += v*v; }
  __shared__ float sh1[256], sh2[256];
  sh1[threadIdx.x] = s; sh2[threadIdx.x] = s2;
  __syncthreads();
  for (int w = 128; w > 0; w >>= 1) {
    if (threadIdx.x < (unsigned)w) { sh1[threadIdx.x] += sh1[threadIdx.x + w];
                                     sh2[threadIdx.x] += sh2[threadIdx.x + w]; }
    __syncthreads();
  }
  if (threadIdx.x == 0) {
    float mean = sh1[0]/(float)n, var = sh2[0]/(float)n - mean*mean;
    float sigma = ws[OFF_SIG + ch*16 + b];
    float scale = (ch == 0) ? 1.0f : 0.5f;
    float sig_y = sqrtf(fmaxf(var - sigma*sigma, 0.0f));
    float t_univ  = scale*sigma*sqrtf(2.0f*logf((float)n));
    float t_bayes = scale*sigma*sigma/fmaxf(sig_y, 1e-10f);
    ws[OFF_THR + ch*96 + band*16 + b] = (sig_y > 1e-10f) ? t_bayes : t_univ;
  }
}

// ---------------- Pass 5: threshold + level-2 IDWT (-> rec1 in ll1 buffer) ----------------
__global__ __launch_bounds__(128) void k_inv2(float* __restrict__ ws) {
  __shared__ float sh[4][2*16*PITCH];
  int wave = threadIdx.x >> 5, lane = threadIdx.x & 31;
  int tile = blockIdx.x*4 + wave;                 // 12288 tiles
  int ch = tile >> 12, rem = tile & 4095;
  int b = rem >> 8, tI = (rem >> 4) & 15, tJ = rem & 15;
  float* xs = &sh[wave][0];
  float* ts = &sh[wave][16*PITCH];
  #pragma unroll
  for (int i2 = 0; i2 < 8; ++i2) {
    int q = lane + 32*i2, r = q >> 4, c = q & 15;
    int region = (r >= 8) ? ((c >= 8) ? 3 : 1) : ((c >= 8) ? 2 : 0);
    const float* srcp = ws + OFF_L2 + (size_t)region*3*S2 + (size_t)ch*S2 + (size_t)b*NQUAR*NQUAR;
    float v = srcp[(size_t)(tI*8 + (r & 7))*NQUAR + (size_t)(tJ*8 + (c & 7))];
    if (region != 0) v = soft_thr(v, ws[OFF_THR + ch*96 + (region + 2)*16 + b]);
    xs[r*PITCH + c] = v;
  }
  asm volatile("s_wait_dscnt 0x0" ::: "memory");
  v16h a1, a2; build_a(lane, 1, a1, a2);          // inverse: A^T Y A
  v8f rr = tile_mxmt(xs, ts, lane, a1, a2);
  int i = lane & 15, mb = (lane >> 4) << 3;
  float* dp = ws + (size_t)ch*S1 + (size_t)b*NHALF*NHALF
                 + (size_t)(tI*16 + i)*NHALF + (size_t)(tJ*16 + mb);
  float4* d4 = (float4*)dp;
  d4[0] = make_float4(rr[0], rr[1], rr[2], rr[3]);
  d4[1] = make_float4(rr[4], rr[5], rr[6], rr[7]);
}

// ---------------- Pass 6: threshold + level-1 IDWT + YCbCr->RGB + clip ----------------
__global__ __launch_bounds__(128) void k_inv1(float* __restrict__ ws, float* __restrict__ out) {
  __shared__ float sh[4][2*16*PITCH];
  int wave = threadIdx.x >> 5, lane = threadIdx.x & 31;
  int tile = blockIdx.x*4 + wave;                 // 16384 tiles
  int b = tile >> 10, tI = (tile >> 5) & 31, tJ = tile & 31;
  float* xs = &sh[wave][0];
  float* ts = &sh[wave][16*PITCH];
  v16h a1, a2; build_a(lane, 1, a1, a2);
  float ycc[3][8];
  #pragma unroll
  for (int ch = 0; ch < 3; ++ch) {
    #pragma unroll
    for (int i2 = 0; i2 < 8; ++i2) {
      int q = lane + 32*i2, r = q >> 4, c = q & 15;
      int region = (r >= 8) ? ((c >= 8) ? 3 : 1) : ((c >= 8) ? 2 : 0);
      const float* srcp = ws + (size_t)region*3*S1 + (size_t)ch*S1 + (size_t)b*NHALF*NHALF;
      float v = srcp[(size_t)(tI*8 + (r & 7))*NHALF + (size_t)(tJ*8 + (c & 7))];
      if (region != 0) v = soft_thr(v, ws[OFF_THR + ch*96 + (region - 1)*16 + b]);
      xs[r*PITCH + c] = v;
    }
    asm volatile("s_wait_dscnt 0x0" ::: "memory");
    v8f rr = tile_mxmt(xs, ts, lane, a1, a2);
    #pragma unroll
    for (int r = 0; r < 8; ++r) ycc[ch][r] = rr[r];
  }
  int i = lane & 15, mb = (lane >> 4) << 3;
  float o[24];
  #pragma unroll
  for (int r = 0; r < 8; ++r) {
    float y = ycc[0][r], cb = ycc[1][r] - 0.5f, cr = ycc[2][r] - 0.5f;
    float R  = y + 1.403f*cr;
    float G  = y - 0.714f*cr - 0.344f*cb;
    float Bl = y + 1.773f*cb;
    o[3*r+0] = fminf(fmaxf(R,  0.0f), 1.0f);
    o[3*r+1] = fminf(fmaxf(G,  0.0f), 1.0f);
    o[3*r+2] = fminf(fmaxf(Bl, 0.0f), 1.0f);
  }
  size_t base = (((size_t)b*NPIX + (size_t)(tI*16 + i))*NPIX + (size_t)(tJ*16 + mb))*3;
  float4* op = (float4*)(out + base);
  #pragma unroll
  for (int q4 = 0; q4 < 6; ++q4)
    op[q4] = make_float4(o[4*q4], o[4*q4+1], o[4*q4+2], o[4*q4+3]);
}

extern "C" void kernel_launch(void* const* d_in, const int* in_sizes, int n_in,
                              void* d_out, int out_size, void* d_ws, size_t ws_size,
                              hipStream_t stream) {
  (void)in_sizes; (void)n_in; (void)out_size; (void)ws_size;
  const float* img = (const float*)d_in[0];
  float* out = (float*)d_out;
  float* ws  = (float*)d_ws;
  k_fwd1 <<<4096, 128, 0, stream>>>(img, ws);
  k_fwd2 <<<3072, 128, 0, stream>>>(ws);
  k_median<<<48,  256, 0, stream>>>(ws);
  k_var  <<<288,  256, 0, stream>>>(ws);
  k_inv2 <<<3072, 128, 0, stream>>>(ws);
  k_inv1 <<<4096, 128, 0, stream>>>(ws, out);
}
